// Decoder_59459527246144
// MI455X (gfx1250) — matrix operand order repaired and checked by reading, not compiled, not run on hardware
//
#include <hip/hip_runtime.h>
#include <hip/hip_bf16.h>

// Problem constants (reference: B=32, S=128, H=256, V=32000, T=100)
constexpr int BB = 32;
constexpr int SS = 128;
constexpr int HH = 256;
constexpr int VV = 32000;
constexpr int TT = 100;

typedef __bf16 bf16_t;
typedef __attribute__((ext_vector_type(16))) __bf16 v16bf;
typedef __attribute__((ext_vector_type(8)))  __bf16 v8bf;
typedef __attribute__((ext_vector_type(8)))  float  v8f;

__device__ __forceinline__ bf16_t f32_to_bf16(float f) {
  unsigned int u = __builtin_bit_cast(unsigned int, f);
  unsigned int r = (u + 0x7FFFu + ((u >> 16) & 1u)) >> 16;   // RNE
  return __builtin_bit_cast(bf16_t, (unsigned short)r);
}

// ---------------------------------------------------------------------------
// Prologue 1: enc_k = enc_outputs @ Kw.T + Kb   (B,S,H); computed once.
// grid = B*S blocks, H threads
__global__ void enck_kernel(const float* __restrict__ enc_out,
                            const float* __restrict__ Kw,
                            const float* __restrict__ Kb,
                            float* __restrict__ enc_k) {
  __shared__ float row[HH];
  const int bs = blockIdx.x;
  const int i  = threadIdx.x;
  row[i] = enc_out[(size_t)bs * HH + i];
  __syncthreads();
  const float* w = Kw + (size_t)i * HH;
  float acc = Kb[i];
  for (int k = 0; k < HH; ++k) acc = fmaf(row[k], w[k], acc);
  enc_k[(size_t)bs * HH + i] = acc;
}

// ---------------------------------------------------------------------------
// Prologue 2: WT[k][v] = bf16(out_w[v][k])  -- K-major bf16 weights (L2-resident)
// grid = V/256 blocks, 256 threads
__global__ void wcvt_kernel(const float* __restrict__ out_w,
                            bf16_t* __restrict__ WT) {
  const int v = blockIdx.x * 256 + threadIdx.x;
  const float* src = out_w + (size_t)v * HH;
  for (int k = 0; k < HH; ++k)
    WT[(size_t)k * VV + v] = f32_to_bf16(src[k]);
}

// ---------------------------------------------------------------------------
// Prologue 3: h0 = enc_hidden[0]
__global__ void hinit_kernel(const float* __restrict__ enc_hidden,
                             float* __restrict__ h0) {
  const int i = blockIdx.x * 256 + threadIdx.x;
  if (i < BB * HH) h0[i] = enc_hidden[i];
}

// ---------------------------------------------------------------------------
// Step kernel A1: embedding gather + additive attention.
// grid = B blocks (one batch each), H=256 threads
__global__ void attn_kernel(int t,
                            const float* __restrict__ hin,        // (B,H)
                            const int*   __restrict__ target,     // (B,T)
                            const float* __restrict__ embedding,  // (V,H)
                            const float* __restrict__ Qw,
                            const float* __restrict__ Qb,
                            const float* __restrict__ Vw,         // (H)
                            const float* __restrict__ Vb,         // (1)
                            const float* __restrict__ enc_k,      // (B,S,H)
                            const float* __restrict__ enc_out,    // (B,S,H)
                            float* __restrict__ xbuf,             // (B,2H) [emb|ctx]
                            float* __restrict__ atts_out) {       // d_out atts
  __shared__ float hs[HH];
  __shared__ float qs[HH];
  __shared__ float partial[SS][2];
  __shared__ float sc[SS];
  __shared__ float wv[SS];
  __shared__ float red[SS];

  const int b = blockIdx.x, tid = threadIdx.x;
  hs[tid] = hin[b * HH + tid];
  const int tok = (t == 0) ? 0 : target[b * TT + (t - 1)];
  xbuf[(size_t)b * 2 * HH + tid] = embedding[(size_t)tok * HH + tid];
  __syncthreads();

  // q = h @ Qw.T + Qb (one output per thread)
  {
    const float* w = Qw + (size_t)tid * HH;
    float acc = Qb[tid];
    for (int k = 0; k < HH; ++k) acc = fmaf(hs[k], w[k], acc);
    qs[tid] = acc;
  }
  __syncthreads();

  // scores[s] = Vw . tanh(q + enc_k[b,s,:]) + Vb   (2 threads per s)
  {
    const int s = tid >> 1, half = tid & 1;
    const float* ek = enc_k + ((size_t)b * SS + s) * HH + half * 128;
    const float* q  = qs + half * 128;
    const float* vw = Vw + half * 128;
    float acc = 0.f;
    for (int k = 0; k < 128; ++k) acc = fmaf(tanhf(q[k] + ek[k]), vw[k], acc);
    partial[s][half] = acc;
  }
  __syncthreads();
  if (tid < SS) { sc[tid] = partial[tid][0] + partial[tid][1] + Vb[0]; red[tid] = sc[tid]; }
  __syncthreads();
  for (int o = 64; o > 0; o >>= 1) {
    if (tid < o) red[tid] = fmaxf(red[tid], red[tid + o]);
    __syncthreads();
  }
  const float mx = red[0];
  __syncthreads();
  if (tid < SS) { float e = __expf(sc[tid] - mx); wv[tid] = e; red[tid] = e; }
  __syncthreads();
  for (int o = 64; o > 0; o >>= 1) {
    if (tid < o) red[tid] += red[tid + o];
    __syncthreads();
  }
  const float inv = 1.f / red[0];
  __syncthreads();
  if (tid < SS) {
    float w = wv[tid] * inv;
    wv[tid] = w;
    atts_out[((size_t)b * TT + t) * SS + tid] = w;
  }
  __syncthreads();

  // ctx = w @ enc_outputs[b]  (one H-component per thread; coalesced per s)
  float acc = 0.f;
  for (int s = 0; s < SS; ++s)
    acc = fmaf(wv[s], enc_out[((size_t)b * SS + s) * HH + tid], acc);
  xbuf[(size_t)b * 2 * HH + HH + tid] = acc;
}

// ---------------------------------------------------------------------------
// Step kernel A2: GRU cell -> h_new (fp32 + bf16 copy for WMMA), h_final slot.
// grid = B blocks, H=256 threads (thread j -> h_new[b][j])
__global__ void gru_kernel(int t,
                           const float* __restrict__ hin,
                           const float* __restrict__ xbuf,   // (B,2H)
                           const float* __restrict__ W_ih,   // (3H,2H)
                           const float* __restrict__ W_hh,   // (3H,H)
                           const float* __restrict__ b_ih,
                           const float* __restrict__ b_hh,
                           float* __restrict__ hout,
                           bf16_t* __restrict__ hbf,
                           float* __restrict__ hfinal) {
  __shared__ float xs[2 * HH];
  __shared__ float hsv[HH];
  const int b = blockIdx.x, j = threadIdx.x;
  xs[j]        = xbuf[(size_t)b * 2 * HH + j];
  xs[j + HH]   = xbuf[(size_t)b * 2 * HH + HH + j];
  hsv[j]       = hin[b * HH + j];
  __syncthreads();

  float gi[3], gh[3];
  for (int g = 0; g < 3; ++g) {
    const int r = g * HH + j;
    const float* wi = W_ih + (size_t)r * 2 * HH;
    float a = b_ih[r];
    for (int k = 0; k < 2 * HH; ++k) a = fmaf(xs[k], wi[k], a);
    gi[g] = a;
    const float* wh = W_hh + (size_t)r * HH;
    float c = b_hh[r];
    for (int k = 0; k < HH; ++k) c = fmaf(hsv[k], wh[k], c);
    gh[g] = c;
  }
  const float r = 1.f / (1.f + __expf(-(gi[0] + gh[0])));
  const float z = 1.f / (1.f + __expf(-(gi[1] + gh[1])));
  const float n = tanhf(gi[2] + r * gh[2]);
  const float hn = (1.f - z) * n + z * hsv[j];
  hout[b * HH + j]   = hn;
  hbf[b * HH + j]    = f32_to_bf16(hn);
  hfinal[b * HH + j] = hn;   // last step's write is the returned h_final
}

// ---------------------------------------------------------------------------
// Step kernel B: logits = h_new(32x256,bf16) @ WT(256x32000,bf16) + out_b
// via v_wmma_f32_16x16x32_bf16. One wave per 16-wide N tile, both M tiles
// (B fragment reused). grid = 500 blocks x 128 threads (4 waves).
__global__ void logits_wmma_kernel(int t,
                                   const bf16_t* __restrict__ hB,   // (32,256)
                                   const bf16_t* __restrict__ WT,   // (256,V) K-major
                                   const float*  __restrict__ out_b,
                                   float* __restrict__ dec) {       // (B,T,V)
  const int lane  = threadIdx.x & 31;
  const int wave  = threadIdx.x >> 5;
  const int ntile = blockIdx.x * 4 + wave;     // 0..1999
  const int n0    = ntile * 16;
  const int lmod  = lane & 15;
  const bool hi   = lane >= 16;

  // A-fragment K offsets within each 32-wide K block (ISA 16-bit A layout)
  const int kA0 = hi ? 8 : 0;        // elements 0..7 -> K = kA0..kA0+7
  const int kA1 = kA0 + 16;          // elements 8..15 -> K = kA1..kA1+7
  // B-fragment: lane -> N column, elements -> same K offsets as A

  v8f acc0 = {};   // M rows 0..15
  v8f acc1 = {};   // M rows 16..31

  for (int kb = 0; kb < HH; kb += 32) {
    v16bf bfrag;
#pragma unroll
    for (int e = 0; e < 8; ++e) {
      bfrag[e]     = WT[(size_t)(kb + kA0 + e) * VV + n0 + lmod];
      bfrag[e + 8] = WT[(size_t)(kb + kA1 + e) * VV + n0 + lmod];
    }

    const bf16_t* p0 = hB + (size_t)lmod * HH + kb;
    const bf16_t* p1 = hB + (size_t)(lmod + 16) * HH + kb;
    v8bf a0lo = *(const v8bf*)(p0 + kA0);
    v8bf a0hi = *(const v8bf*)(p0 + kA1);
    v8bf a1lo = *(const v8bf*)(p1 + kA0);
    v8bf a1hi = *(const v8bf*)(p1 + kA1);
    v16bf a0, a1;
#pragma unroll
    for (int e = 0; e < 8; ++e) {
      a0[e] = a0lo[e]; a0[e + 8] = a0hi[e];
      a1[e] = a1lo[e]; a1[e + 8] = a1hi[e];
    }
    acc0 = __builtin_amdgcn_wmma_f32_16x16x32_bf16(false, a0, false, bfrag,
                                                   (short)0, acc0, false, false);
    acc1 = __builtin_amdgcn_wmma_f32_16x16x32_bf16(false, a1, false, bfrag,
                                                   (short)0, acc1, false, false);
  }

  // D layout: lane -> N, VGPR e -> M row (e, or e+8 for high lanes)
  const int n = n0 + lmod;
  const float bias = out_b[n];
#pragma unroll
  for (int e = 0; e < 8; ++e) {
    const int m0 = e + (hi ? 8 : 0);
    dec[((size_t)m0 * TT + t) * VV + n]        = acc0[e] + bias;
    dec[((size_t)(m0 + 16) * TT + t) * VV + n] = acc1[e] + bias;
  }
}

// ---------------------------------------------------------------------------
// Step kernel C: in-place log_softmax over V for row (b, t).
// grid = B blocks, 256 threads
__global__ void lsm_kernel(int t, float* __restrict__ dec) {
  __shared__ float red[256];
  const int b = blockIdx.x, tid = threadIdx.x;
  float* row = dec + ((size_t)b * TT + t) * VV;

  float mx = -3.4e38f;
  for (int v = tid; v < VV; v += 256) mx = fmaxf(mx, row[v]);
  red[tid] = mx; __syncthreads();
  for (int o = 128; o > 0; o >>= 1) {
    if (tid < o) red[tid] = fmaxf(red[tid], red[tid + o]);
    __syncthreads();
  }
  mx = red[0];
  __syncthreads();

  float s = 0.f;
  for (int v = tid; v < VV; v += 256) s += __expf(row[v] - mx);
  red[tid] = s; __syncthreads();
  for (int o = 128; o > 0; o >>= 1) {
    if (tid < o) red[tid] += red[tid + o];
    __syncthreads();
  }
  const float lz = mx + __logf(red[0]);
  for (int v = tid; v < VV; v += 256) row[v] -= lz;
}

// ---------------------------------------------------------------------------
extern "C" void kernel_launch(void* const* d_in, const int* in_sizes, int n_in,
                              void* d_out, int out_size, void* d_ws, size_t ws_size,
                              hipStream_t stream) {
  const float* enc_outputs = (const float*)d_in[0];
  const float* enc_hidden  = (const float*)d_in[1];
  const int*   target      = (const int*)  d_in[2];
  const float* embedding   = (const float*)d_in[3];
  const float* Qw   = (const float*)d_in[4];
  const float* Qb   = (const float*)d_in[5];
  const float* Kw   = (const float*)d_in[6];
  const float* Kb   = (const float*)d_in[7];
  const float* Vw   = (const float*)d_in[8];
  const float* Vb   = (const float*)d_in[9];
  const float* W_ih = (const float*)d_in[10];
  const float* W_hh = (const float*)d_in[11];
  const float* b_ih = (const float*)d_in[12];
  const float* b_hh = (const float*)d_in[13];
  const float* out_w = (const float*)d_in[14];
  const float* out_b = (const float*)d_in[15];

  float* dec    = (float*)d_out;                        // (B,T,V)
  float* hfinal = dec + (size_t)BB * TT * VV;           // (1,B,H)
  float* atts   = hfinal + (size_t)BB * HH;             // (B,T,S)

  // Workspace layout (256B aligned): enc_k | WT(bf16) | h0 | h1 | hbf | xbuf
  char* ws = (char*)d_ws;
  float*  enc_k = (float*)(ws);                                      // 4,194,304 B
  bf16_t* WT    = (bf16_t*)(ws + 4194304);                           // 16,384,000 B
  float*  h0    = (float*)(ws + 20578304);                           // 32,768 B
  float*  h1    = (float*)(ws + 20611072);                           // 32,768 B
  bf16_t* hbf   = (bf16_t*)(ws + 20643840);                          // 16,384 B
  float*  xbuf  = (float*)(ws + 20660224);                           // 65,536 B

  // Prologue (step-invariant work)
  enck_kernel<<<BB * SS, HH, 0, stream>>>(enc_outputs, Kw, Kb, enc_k);
  wcvt_kernel<<<VV / 256, 256, 0, stream>>>(out_w, WT);
  hinit_kernel<<<(BB * HH + 255) / 256, 256, 0, stream>>>(enc_hidden, h0);

  for (int t = 0; t < TT; ++t) {
    float* hin  = (t & 1) ? h1 : h0;
    float* hout = (t & 1) ? h0 : h1;
    attn_kernel<<<BB, HH, 0, stream>>>(t, hin, target, embedding, Qw, Qb,
                                       Vw, Vb, enc_k, enc_outputs, xbuf, atts);
    gru_kernel<<<BB, HH, 0, stream>>>(t, hin, xbuf, W_ih, W_hh, b_ih, b_hh,
                                      hout, hbf, hfinal);
    logits_wmma_kernel<<<(VV / 16) / 4, 128, 0, stream>>>(t, hbf, WT, out_b, dec);
    lsm_kernel<<<BB, 256, 0, stream>>>(t, dec);
  }
}
